// EncoderAttention_7730941133121
// MI455X (gfx1250) — matrix operand order repaired and checked by reading, not compiled, not run on hardware
//
#include <hip/hip_runtime.h>
#include <hip/hip_bf16.h>

typedef __attribute__((ext_vector_type(16))) __bf16 v16bf;
typedef __attribute__((ext_vector_type(8)))  float  v8f;

#define H2   1024
#define TLEN 2048
#define NB   32
#define LDSW (H2 + 8)   // padded LDS row stride (bf16 elems) -> bank-conflict-free

static __device__ __forceinline__ unsigned int pack2bf(float a, float b) {
    union { __bf16 h[2]; unsigned int u; } p;
    p.h[0] = (__bf16)a;
    p.h[1] = (__bf16)b;
    return p.u;
}

// branch-free tanh: 1 - 2/(e^{2x}+1); clamp keeps exp finite (tanh(|10|)==1 in fp32)
static __device__ __forceinline__ float tanh_fast(float x) {
    float xc = fminf(fmaxf(x, -10.f), 10.f);
    float t  = __expf(2.f * xc);                    // v_exp_f32
    return 1.f - 2.f * __builtin_amdgcn_rcpf(t + 1.f); // v_rcp_f32
}

// ---------------- K0: convert Wh (1024x1024 fp32) -> bf16 in workspace ----
__global__ __launch_bounds__(256) void k_cvt_wh(const float* __restrict__ wh,
                                                unsigned int* __restrict__ whb) {
    int i4 = blockIdx.x * 256 + threadIdx.x;      // each thread: 4 elements
    float4 f = ((const float4*)wh)[i4];
    uint2 p;
    p.x = pack2bf(f.x, f.y);
    p.y = pack2bf(f.z, f.w);
    ((uint2*)whb)[i4] = p;
}

// ---------------- K1: dec[b,e] = DH[b,:] . Ws[e,:] + bs[e] ----------------
__global__ __launch_bounds__(256) void k_dec(const float* __restrict__ dh,
                                             const float* __restrict__ wsm,
                                             const float* __restrict__ bs,
                                             float* __restrict__ dec) {
    int idx = blockIdx.x * 256 + threadIdx.x;
    int b = idx & 31;            // 32 lanes of a wave share one Ws row
    int e = idx >> 5;
    const float4* dhr = (const float4*)(dh + (size_t)b * H2);
    const float4* wsr = (const float4*)(wsm + (size_t)e * H2);
    float acc = bs[e];
    for (int d = 0; d < H2 / 4; ++d) {
        float4 a = dhr[d];
        float4 w = wsr[d];
        acc += a.x * w.x + a.y * w.y + a.z * w.z + a.w * w.w;
    }
    dec[b * H2 + e] = acc;
}

// ---------------- K2: fused bf16-WMMA GEMM + tanh + v-dot -> scores -------
// One block = 16 rows of (b,t). 8 waves; wave w owns one 128-column N-group:
// 8 independent WMMA chains off a single B base pointer (immediate offsets).
__global__ __launch_bounds__(256) void k_scores(const float* __restrict__ eh,
                                                const unsigned int* __restrict__ whb,
                                                const float* __restrict__ dec,
                                                const float* __restrict__ vvec,
                                                float* __restrict__ scores) {
    __shared__ __align__(16) unsigned int ldsA[16 * LDSW / 2];   // 16x1024 bf16, padded rows
    __shared__ float sred[16];

    int tid = threadIdx.x;
    int b   = blockIdx.x >> 7;
    int t0  = (blockIdx.x & 127) << 4;

    // ---- stage A tile (16x1024 fp32 -> bf16 LDS), fully coalesced --------
    const float4* src = (const float4*)(eh + ((size_t)b * TLEN + t0) * H2);
    uint2* dst = (uint2*)ldsA;
#pragma unroll
    for (int j = 0; j < 16; ++j) {
        int i4   = tid + j * 256;          // float4 index within contiguous tile
        int lin  = i4 * 4;
        int row  = lin >> 10;              // /1024
        int col  = lin & 1023;
        float4 f = src[i4];
        uint2 p;
        p.x = pack2bf(f.x, f.y);
        p.y = pack2bf(f.z, f.w);
        dst[(row * LDSW + col) >> 1] = p;
    }
    if (tid < 16) sred[tid] = 0.f;
    __syncthreads();

    int w     = tid >> 5;
    int lane  = tid & 31;
    int r     = lane & 15;
    int khalf = lane >> 4;

    const uint4* lA = (const uint4*)ldsA;
    const uint4* B4 = (const uint4*)whb;

    int nbase = w * 128 + r;               // this lane's column within each 16-chunk
    // single per-lane B base; chunk ch is a +ch*32KB immediate offset
    const uint4* bp = B4 + (((size_t)nbase * H2 + khalf * 16) >> 3);

    v8f c[8] = {v8f{}, v8f{}, v8f{}, v8f{}, v8f{}, v8f{}, v8f{}, v8f{}};

    for (int k0 = 0; k0 < H2; k0 += 32) {
        // A fragment (lane r, K = kh*16 .. kh*16+15, contiguous)
        union { uint4 u[2]; v16bf v; } a;
        int ea  = r * LDSW + k0 + khalf * 16;
        a.u[0] = lA[ea >> 3];
        a.u[1] = lA[(ea + 8) >> 3];

#pragma unroll
        for (int ch = 0; ch < 8; ++ch) {
            union { uint4 u[2]; v16bf v; } bf;
            bf.u[0] = bp[ch * 2048];       // 16*H2/8 uint4 = 32KB row-block stride
            bf.u[1] = bp[ch * 2048 + 1];
            c[ch] = __builtin_amdgcn_wmma_f32_16x16x32_bf16(
                false, a.v, false, bf.v, (short)0, c[ch], false, false);
        }
        bp += 4;                            // advance K by 32 bf16 elems
    }

    // epilogue: tanh + dot with v (C layout: lane<16 -> rows 0-7, else 8-15)
    float sacc[8];
#pragma unroll
    for (int i = 0; i < 8; ++i) sacc[i] = 0.f;
#pragma unroll
    for (int ch = 0; ch < 8; ++ch) {
        int n    = nbase + ch * 16;
        float dv = dec[b * H2 + n];
        float vv = vvec[n];
#pragma unroll
        for (int g8 = 0; g8 < 8; ++g8)
            sacc[g8] += vv * tanh_fast(c[ch][g8] + dv);
    }

    // reduce over the 16 lanes of each half (keeps row halves separate)
#pragma unroll
    for (int m = 8; m >= 1; m >>= 1) {
#pragma unroll
        for (int g8 = 0; g8 < 8; ++g8)
            sacc[g8] += __shfl_xor(sacc[g8], m, 32);
    }
    if (r == 0) {
        int rowbase = khalf * 8;
#pragma unroll
        for (int g8 = 0; g8 < 8; ++g8)
            atomicAdd(&sred[rowbase + g8], sacc[g8]);
    }
    __syncthreads();
    if (tid < 16) scores[(size_t)b * TLEN + t0 + tid] = sred[tid];
}

// ---------------- K3: exp-normalize, mask, softmax, context ---------------
__global__ __launch_bounds__(256) void k_final(const float* __restrict__ eh,
                                               const float* __restrict__ pad,
                                               const float* __restrict__ sumt,
                                               const float* __restrict__ scores,
                                               float* __restrict__ o_cte,
                                               float* __restrict__ o_at,
                                               float* __restrict__ o_stn) {
    __shared__ float at_lds[TLEN];
    __shared__ float red[8];
    int b   = blockIdx.x;
    int tid = threadIdx.x;

    float psum = 0.f;
    for (int t = tid; t < TLEN; t += 256) {
        size_t o = (size_t)b * TLEN + t;
        float e  = __expf(scores[o]);
        float st = sumt[o];
        o_stn[o] = st + e;
        float a  = e * __builtin_amdgcn_rcpf(st) * pad[o];
        at_lds[t] = a;
        psum += a;
    }
#pragma unroll
    for (int m = 16; m >= 1; m >>= 1) psum += __shfl_xor(psum, m, 32);
    if ((tid & 31) == 0) red[tid >> 5] = psum;
    __syncthreads();

    float tot = 0.f;
#pragma unroll
    for (int i = 0; i < 8; ++i) tot += red[i];
    float inv = __builtin_amdgcn_rcpf(tot);

    for (int t = tid; t < TLEN; t += 256) {
        float a   = at_lds[t] * inv;
        at_lds[t] = a;
        o_at[(size_t)b * TLEN + t] = a;
    }
    __syncthreads();

    // cte[b,d] = sum_t at[t] * EH[b,t,d] ; threads cover d -> coalesced rows
    for (int d = tid; d < H2; d += 256) {
        float acc = 0.f;
        const float* col = eh + (size_t)b * TLEN * H2 + d;
#pragma unroll 8
        for (int t = 0; t < TLEN; ++t)
            acc += at_lds[t] * col[(size_t)t * H2];
        o_cte[b * H2 + d] = acc;
    }
}

extern "C" void kernel_launch(void* const* d_in, const int* in_sizes, int n_in,
                              void* d_out, int out_size, void* d_ws, size_t ws_size,
                              hipStream_t stream) {
    const float* dh   = (const float*)d_in[0];   // decoder_hidden [32,1024]
    const float* eh   = (const float*)d_in[1];   // encoder_hidden [32,2048,1024]
    const float* pad  = (const float*)d_in[2];   // encoder_padding [32,2048]
    const float* sumt = (const float*)d_in[3];   // sum_temporal [32,2048]
    const float* Wh   = (const float*)d_in[4];   // [1024,1024]
    const float* Wsm  = (const float*)d_in[5];   // [1024,1024]
    const float* bs   = (const float*)d_in[6];   // [1024]
    const float* v    = (const float*)d_in[7];   // [1024]

    float* out   = (float*)d_out;
    float* o_cte = out;                       // [32,1024]
    float* o_at  = out + NB * H2;             // [32,2048]
    float* o_stn = o_at + NB * TLEN;          // [32,2048]

    float* ws_dec          = (float*)d_ws;                    // 32*1024
    float* ws_scores       = ws_dec + NB * H2;                // 32*2048
    unsigned int* ws_whb   = (unsigned int*)(ws_scores + NB * TLEN); // 1M bf16 = 2MB

    k_cvt_wh<<<(H2 * H2) / (256 * 4), 256, 0, stream>>>(Wh, ws_whb);
    k_dec<<<(NB * H2) / 256, 256, 0, stream>>>(dh, Wsm, bs, ws_dec);
    k_scores<<<(NB * TLEN) / 16, 256, 0, stream>>>(eh, ws_whb, ws_dec, v, ws_scores);
    k_final<<<NB, 256, 0, stream>>>(eh, pad, sumt, ws_scores, o_cte, o_at, o_stn);
}
